// lineConGraph_57269093925539
// MI455X (gfx1250) — matrix operand order repaired and checked by reading, not compiled, not run on hardware
//
#include <hip/hip_runtime.h>
#include <hip/hip_fp16.h>

typedef __attribute__((ext_vector_type(16))) _Float16 v16h;
typedef __attribute__((ext_vector_type(8)))  _Float16 v8h;
typedef __attribute__((ext_vector_type(8)))  float    v8f;

#define HEADS 4
#define CH    64
#define FEAT  256   // HEADS*CH
#define GK    256   // K dim of every GEMM in this net
#define BCOLS 64    // output columns per block (one LDS weight panel)
#define LDS_STRIDE (GK + 8)   // halfs; 16B pad -> bank-group spread for ds_load_b128

// ---------- helpers ----------
__device__ __forceinline__ unsigned f2key(float f) {
  unsigned b = __float_as_uint(f);
  return (b & 0x80000000u) ? ~b : (b | 0x80000000u);   // monotonic order-preserving key
}
__device__ __forceinline__ float key2f(unsigned k) {
  unsigned b = (k & 0x80000000u) ? (k ^ 0x80000000u) : ~k;
  return __uint_as_float(b);
}
__device__ __forceinline__ float lrelu(float v) { return v > 0.f ? v : 0.2f * v; }
__device__ __forceinline__ float eluf(float v)  { return v > 0.f ? v : expm1f(v); }

__device__ __forceinline__ void async_b128_to_lds(const void* gptr, const void* lptr) {
  unsigned lds = (unsigned)(uintptr_t)lptr;                 // LDS aperture: addr[31:0]
  unsigned long long g = (unsigned long long)(uintptr_t)gptr;
  asm volatile("global_load_async_to_lds_b128 %0, %1, off" :: "v"(lds), "v"(g) : "memory");
}
__device__ __forceinline__ void wait_async0() {
#if __has_builtin(__builtin_amdgcn_s_wait_asynccnt)
  __builtin_amdgcn_s_wait_asynccnt(0);
#else
  asm volatile("s_wait_asynccnt 0x0" ::: "memory");
#endif
}

// ---------- conversions ----------
__global__ void cvt_f32_to_f16(const float* __restrict__ src, _Float16* __restrict__ dst, int n) {
  int i = blockIdx.x * blockDim.x + threadIdx.x;
  if (i < n) dst[i] = (_Float16)src[i];
}

// Wt[n*K + k] = (f16) W[k*N + n]   (weights -> column-major f16 for contiguous B fragments)
__global__ void cvt_transpose_f16(const float* __restrict__ W, _Float16* __restrict__ Wt, int K, int N) {
  int o = blockIdx.x * blockDim.x + threadIdx.x;
  if (o >= K * N) return;
  int n = o / K, k = o % K;
  Wt[o] = (_Float16)W[(size_t)k * N + n];
}

// ---------- WMMA GEMM: C[M,N] = A[M,GK] (f16 row-major) x B (as Bt[N,GK] f16) ----------
// Block = 8 waves; block owns 128 rows x 64 cols of C and stages its 64xGK weight
// panel into LDS once via gfx1250 async global->LDS DMA (ASYNCcnt), shared by all waves.
__global__ __launch_bounds__(256)
void gemm_wmma_f16(const _Float16* __restrict__ A, const _Float16* __restrict__ Bt,
                   float* __restrict__ C, int M, int N, int nCC) {
  __shared__ _Float16 Bs[BCOLS * LDS_STRIDE];
  const int blockRow = blockIdx.x / nCC;
  const int colChunk = blockIdx.x % nCC;

  // ---- async-stage weight panel: 64 x 256 f16 = 2048 x 16B units ----
  {
    const _Float16* gbase = Bt + (size_t)colChunk * BCOLS * GK;
#pragma unroll
    for (int i = 0; i < 8; ++i) {
      int u  = threadIdx.x + 256 * i;   // unit id
      int n  = u >> 5;                  // 32 units per panel row
      int k8 = u & 31;
      async_b128_to_lds(gbase + (size_t)n * GK + k8 * 8, &Bs[n * LDS_STRIDE + k8 * 8]);
    }
    wait_async0();
    __syncthreads();
  }

  const int wave = threadIdx.x >> 5;
  const int lane = threadIdx.x & 31;
  const int rowTile = blockRow * 8 + wave;
  if (rowTile * 16 >= M) return;                 // uniform per wave: EXEC stays all-ones
  const int mrow = rowTile * 16 + (lane & 15);   // A layout: M = lane&15
  const int hi = lane >> 4;
  const int nl = lane & 15;                      // B/C layout: N = lane&15 (+tile)

  v8f acc[4] = {v8f{}, v8f{}, v8f{}, v8f{}};
  for (int k0 = 0; k0 < GK; k0 += 32) {
    // A fragment: lane holds row mrow, K = [k0+8hi, +8) and [k0+16+8hi, +8)  (two 16B loads)
    const _Float16* ap = A + (size_t)mrow * GK + k0 + 8 * hi;
    v8h alo = *(const v8h*)(ap);
    v8h ahi = *(const v8h*)(ap + 16);
    v16h a;
#pragma unroll
    for (int i = 0; i < 8; ++i) { a[i] = alo[i]; a[8 + i] = ahi[i]; }
#pragma unroll
    for (int t = 0; t < 4; ++t) {
      // B fragment from LDS: lane holds col (nl + t*16), K = [k0+8hi, +8) and [k0+16+8hi, +8)  (two ds_load_b128)
      const _Float16* bp = &Bs[(nl + t * 16) * LDS_STRIDE + k0 + 8 * hi];
      v8h blo = *(const v8h*)(bp);
      v8h bhi = *(const v8h*)(bp + 16);
      v16h b;
#pragma unroll
      for (int i = 0; i < 8; ++i) { b[i] = blo[i]; b[8 + i] = bhi[i]; }
      acc[t] = __builtin_amdgcn_wmma_f32_16x16x32_f16(
          false, a, false, b, (short)0, acc[t], false, false);
    }
  }
#pragma unroll
  for (int t = 0; t < 4; ++t) {
#pragma unroll
    for (int v = 0; v < 8; ++v) {
      int row = rowTile * 16 + 8 * hi + v;       // C layout: M = v + 8*hi
      C[(size_t)row * N + colChunk * 64 + t * 16 + nl] = acc[t][v];
    }
  }
}

// ---------- per-node attention logits: as[n,h] = sum_c h[n,h,c]*a_src[h,c] (and ad) ----------
__global__ void node_alpha_kernel(const float* __restrict__ feat,
                                  const float* __restrict__ a_src, const float* __restrict__ a_dst,
                                  float* __restrict__ as_, float* __restrict__ ad_, int N) {
  int i = blockIdx.x * blockDim.x + threadIdx.x;
  if (i >= N * HEADS) return;
  int n = i >> 2, h = i & 3;
  const float* fp = feat + (size_t)n * FEAT + h * CH;
  float s = 0.f, d = 0.f;
#pragma unroll 8
  for (int c = 0; c < CH; ++c) {
    float v = fp[c];
    s += v * a_src[h * CH + c];
    d += v * a_dst[h * CH + c];
  }
  as_[i] = s; ad_[i] = d;
}

// ---------- edge pass 1: segment max (ordered-uint key atomicMax) ----------
__global__ void edge_max_kernel(const int* __restrict__ src, const int* __restrict__ dst,
                                int E, int N, const float* __restrict__ as_,
                                const float* __restrict__ ad_, unsigned* __restrict__ emaxKey) {
  int e = blockIdx.x * blockDim.x + threadIdx.x;
  if (e >= E + N) return;
  int s, d;
  if (e < E) { s = src[e]; d = dst[e]; } else { s = d = e - E; }
#pragma unroll
  for (int h = 0; h < HEADS; ++h) {
    float v = lrelu(as_[s * HEADS + h] + ad_[d * HEADS + h]);
    atomicMax(&emaxKey[d * HEADS + h], f2key(v));
  }
}

// ---------- edge pass 2: segment sum of exp(e - max) ----------
__global__ void edge_denom_kernel(const int* __restrict__ src, const int* __restrict__ dst,
                                  int E, int N, const float* __restrict__ as_,
                                  const float* __restrict__ ad_,
                                  const unsigned* __restrict__ emaxKey,
                                  float* __restrict__ denom) {
  int e = blockIdx.x * blockDim.x + threadIdx.x;
  if (e >= E + N) return;
  int s, d;
  if (e < E) { s = src[e]; d = dst[e]; } else { s = d = e - E; }
#pragma unroll
  for (int h = 0; h < HEADS; ++h) {
    int di = d * HEADS + h;
    float v = lrelu(as_[s * HEADS + h] + ad_[di]);
    atomicAdd(&denom[di], __expf(v - key2f(emaxKey[di])));
  }
}

// ---------- edge pass 3: agg[d,:] += alpha * feat[s,:]  (64 threads per edge, loop heads) ----------
__global__ void edge_agg_kernel(const int* __restrict__ src, const int* __restrict__ dst,
                                int E, int N, const float* __restrict__ as_,
                                const float* __restrict__ ad_,
                                const unsigned* __restrict__ emaxKey,
                                const float* __restrict__ denom,
                                const float* __restrict__ feat, float* __restrict__ agg) {
  long tid = (long)blockIdx.x * blockDim.x + threadIdx.x;
  int e = (int)(tid >> 6);
  int c = (int)(tid & 63);
  if (e >= E + N) return;
  int s, d;
  if (e < E) { s = src[e]; d = dst[e]; } else { s = d = e - E; }
#pragma unroll
  for (int h = 0; h < HEADS; ++h) {
    int di = d * HEADS + h;
    float v = lrelu(as_[s * HEADS + h] + ad_[di]);
    float alpha = __expf(v - key2f(emaxKey[di])) / (denom[di] + 1e-16f);
    atomicAdd(&agg[(size_t)d * FEAT + h * CH + c],
              feat[(size_t)s * FEAT + h * CH + c] * alpha);
  }
}

// ---------- bias + ELU (in place) ----------
__global__ void bias_elu_kernel(float* __restrict__ x, const float* __restrict__ bias, int n) {
  int i = blockIdx.x * blockDim.x + threadIdx.x;
  if (i < n) x[i] = eluf(x[i] + bias[i & (FEAT - 1)]);
}

// ---------- sum += ELU(x + bias) ----------
__global__ void bias_elu_add_kernel(const float* __restrict__ x, const float* __restrict__ bias,
                                    float* __restrict__ sum, int n) {
  int i = blockIdx.x * blockDim.x + threadIdx.x;
  if (i < n) sum[i] += eluf(x[i] + bias[i & (FEAT - 1)]);
}

// ---------- fused bias + row softmax over 64 classes (in place on d_out) ----------
__global__ void softmax64_kernel(float* __restrict__ out, const float* __restrict__ bfc) {
  __shared__ float red[64];
  int row = blockIdx.x, t = threadIdx.x;
  float v = out[(size_t)row * 64 + t] + bfc[t];
  red[t] = v; __syncthreads();
  for (int s = 32; s > 0; s >>= 1) { if (t < s) red[t] = fmaxf(red[t], red[t + s]); __syncthreads(); }
  float m = red[0]; __syncthreads();
  float e = __expf(v - m);
  red[t] = e; __syncthreads();
  for (int s = 32; s > 0; s >>= 1) { if (t < s) red[t] += red[t + s]; __syncthreads(); }
  out[(size_t)row * 64 + t] = e / red[0];
}

// =================== host orchestration ===================
extern "C" void kernel_launch(void* const* d_in, const int* in_sizes, int n_in,
                              void* d_out, int out_size, void* d_ws, size_t ws_size,
                              hipStream_t stream) {
  const float* x    = (const float*)d_in[0];
  const int*   eo   = (const int*)d_in[1];
  const int*   es   = (const int*)d_in[2];
  const float* W1   = (const float*)d_in[3];
  const float* aS1  = (const float*)d_in[4];
  const float* aD1  = (const float*)d_in[5];
  const float* b1   = (const float*)d_in[6];
  const float* W2   = (const float*)d_in[7];
  const float* aS2  = (const float*)d_in[8];
  const float* aD2  = (const float*)d_in[9];
  const float* b2   = (const float*)d_in[10];
  const float* Wfc  = (const float*)d_in[11];
  const float* bfc  = (const float*)d_in[12];
  float* out = (float*)d_out;

  const int N = in_sizes[0] / FEAT;     // 50000
  const int E = in_sizes[1] / 2;        // 800000
  const int Etot = E + N;
  const size_t NB = (size_t)N * FEAT * sizeof(float);

  char* ws = (char*)d_ws;
  size_t off = 0;
  auto alloc = [&](size_t bytes) { void* p = ws + off; off = (off + bytes + 255) & ~(size_t)255; return p; };

  float*     h1   = (float*)alloc(NB);
  float*     agg  = (float*)alloc(NB);
  float*     h2   = (float*)alloc(NB);
  float*     sum  = (float*)alloc(NB);
  _Float16*  A16  = (_Float16*)alloc((size_t)N * FEAT * sizeof(_Float16));
  _Float16*  W1t  = (_Float16*)alloc((size_t)FEAT * FEAT * sizeof(_Float16));
  _Float16*  W2t  = (_Float16*)alloc((size_t)FEAT * FEAT * sizeof(_Float16));
  _Float16*  Wft  = (_Float16*)alloc((size_t)FEAT * 64 * sizeof(_Float16));
  float*     as1  = (float*)alloc((size_t)N * HEADS * sizeof(float));
  float*     ad1  = (float*)alloc((size_t)N * HEADS * sizeof(float));
  float*     as2  = (float*)alloc((size_t)N * HEADS * sizeof(float));
  float*     ad2  = (float*)alloc((size_t)N * HEADS * sizeof(float));
  unsigned*  emk  = (unsigned*)alloc((size_t)N * HEADS * sizeof(unsigned));
  float*     den  = (float*)alloc((size_t)N * HEADS * sizeof(float));

  const int TB = 256;
  const int gNF = (N * FEAT + TB - 1) / TB;
  const int gNH = (N * HEADS + TB - 1) / TB;
  const int gE  = (Etot + TB - 1) / TB;
  const int gE64 = (int)(((long)Etot * 64 + TB - 1) / TB);

  auto launch_gemm = [&](const _Float16* A, const _Float16* Bt, float* C, int M, int Ncols) {
    int nCC = Ncols / BCOLS;
    int rowTiles = (M + 15) / 16;
    int blocks = ((rowTiles + 7) / 8) * nCC;           // 8 waves/block, 1 panel/block
    gemm_wmma_f16<<<blocks, TB, 0, stream>>>(A, Bt, C, M, Ncols, nCC);
  };

  // ---- one-time prep: weights -> col-major f16, x -> f16, shared layer-1 projection ----
  cvt_transpose_f16<<<(FEAT * FEAT + TB - 1) / TB, TB, 0, stream>>>(W1, W1t, FEAT, FEAT);
  cvt_transpose_f16<<<(FEAT * FEAT + TB - 1) / TB, TB, 0, stream>>>(W2, W2t, FEAT, FEAT);
  cvt_transpose_f16<<<(FEAT * 64 + TB - 1) / TB, TB, 0, stream>>>(Wfc, Wft, FEAT, 64);
  cvt_f32_to_f16<<<gNF, TB, 0, stream>>>(x, A16, N * FEAT);
  launch_gemm(A16, W1t, h1, N, FEAT);                  // h1 = x @ W1 (shared by both branches)
  node_alpha_kernel<<<gNH, TB, 0, stream>>>(h1, aS1, aD1, as1, ad1, N);

  (void)hipMemsetAsync(sum, 0, NB, stream);

  auto run_branch = [&](const int* srcp, const int* dstp) {
    // ---- layer 1 edge softmax + aggregation (uses shared h1/as1/ad1) ----
    (void)hipMemsetAsync(emk, 0, (size_t)N * HEADS * 4, stream);
    (void)hipMemsetAsync(den, 0, (size_t)N * HEADS * 4, stream);
    (void)hipMemsetAsync(agg, 0, NB, stream);
    edge_max_kernel  <<<gE,   TB, 0, stream>>>(srcp, dstp, E, N, as1, ad1, emk);
    edge_denom_kernel<<<gE,   TB, 0, stream>>>(srcp, dstp, E, N, as1, ad1, emk, den);
    edge_agg_kernel  <<<gE64, TB, 0, stream>>>(srcp, dstp, E, N, as1, ad1, emk, den, h1, agg);
    bias_elu_kernel  <<<gNF,  TB, 0, stream>>>(agg, b1, N * FEAT);
    // ---- layer 2 projection ----
    cvt_f32_to_f16<<<gNF, TB, 0, stream>>>(agg, A16, N * FEAT);
    launch_gemm(A16, W2t, h2, N, FEAT);
    node_alpha_kernel<<<gNH, TB, 0, stream>>>(h2, aS2, aD2, as2, ad2, N);
    // ---- layer 2 edge softmax + aggregation (agg reused) ----
    (void)hipMemsetAsync(emk, 0, (size_t)N * HEADS * 4, stream);
    (void)hipMemsetAsync(den, 0, (size_t)N * HEADS * 4, stream);
    (void)hipMemsetAsync(agg, 0, NB, stream);
    edge_max_kernel  <<<gE,   TB, 0, stream>>>(srcp, dstp, E, N, as2, ad2, emk);
    edge_denom_kernel<<<gE,   TB, 0, stream>>>(srcp, dstp, E, N, as2, ad2, emk, den);
    edge_agg_kernel  <<<gE64, TB, 0, stream>>>(srcp, dstp, E, N, as2, ad2, emk, den, h2, agg);
    bias_elu_add_kernel<<<gNF, TB, 0, stream>>>(agg, b2, sum, N * FEAT);
  };

  run_branch(eo, eo + E);   // edge_index_other: row0 = src, row1 = dst
  run_branch(es, es + E);   // edge_idx_speaker

  // ---- final FC + softmax ----
  cvt_f32_to_f16<<<gNF, TB, 0, stream>>>(sum, A16, N * FEAT);
  launch_gemm(A16, Wft, out, N, 64);
  softmax64_kernel<<<N, 64, 0, stream>>>(out, bfc);
}
